// BEVKDV2_22574348108169
// MI455X (gfx1250) — matrix operand order repaired and checked by reading, not compiled, not run on hardware
//
#include <hip/hip_runtime.h>
#include <hip/hip_bf16.h>

typedef __attribute__((ext_vector_type(16))) _Float16 v16h;
typedef __attribute__((ext_vector_type(8)))  float    v8f;
typedef __attribute__((ext_vector_type(4)))  unsigned v4u;

// ---------------------------------------------------------------------------
// Order-preserving float <-> uint encoding so atomicMin/Max on uint implement
// float min/max (works for negatives).
// ---------------------------------------------------------------------------
__device__ __forceinline__ unsigned fenc(float f) {
    unsigned u = __float_as_uint(f);
    return (u >> 31) ? ~u : (u | 0x80000000u);
}
__device__ __forceinline__ float fdec(unsigned u) {
    unsigned v = (u >> 31) ? (u & 0x7fffffffu) : ~u;
    return __uint_as_float(v);
}
__device__ __forceinline__ unsigned short h16(float f) {
    _Float16 h = (_Float16)f;
    return __builtin_bit_cast(unsigned short, h);
}

// ---------------------------------------------------------------------------
// BEV rasterization in place in the NCHW bev tensor:
// plane (b*4 + c), c: 0=count(float) 1=zmin(enc) 2=zmax(enc) 3=imax(enc)
// ---------------------------------------------------------------------------
__global__ void bev_init_kernel(unsigned* __restrict__ bev, int n) {
    int i = blockIdx.x * blockDim.x + threadIdx.x;
    if (i >= n) return;
    int c = (i >> 20) & 3;
    unsigned v;
    if      (c == 0) v = 0u;             // count = 0.0f
    else if (c == 1) v = fenc(10.0f);    // zmin init
    else if (c == 2) v = fenc(-10.0f);   // zmax init
    else             v = fenc(0.0f);     // intensity-max init
    bev[i] = v;
}

__global__ void bev_scatter_kernel(const float* __restrict__ pts,
                                   unsigned* __restrict__ bev, int npts) {
    int i = blockIdx.x * blockDim.x + threadIdx.x;
    if (i >= npts) return;
    float bf = pts[i * 5 + 0];
    float x  = pts[i * 5 + 1];
    float y  = pts[i * 5 + 2];
    float z  = pts[i * 5 + 3];
    float it = pts[i * 5 + 4];
    int xp = (int)(x * (1024.0f / 69.12f));
    int yp = (int)((y + 39.68f) * (1024.0f / 79.36f));
    if (xp < 0 || xp >= 1024 || yp < 0 || yp >= 1024) return;
    unsigned base = ((unsigned)(int)bf * 4u) << 20;
    unsigned pix  = ((unsigned)yp << 10) | (unsigned)xp;
    atomicAdd((float*)(bev + base + pix), 1.0f);
    atomicMin(bev + base + (1u << 20) + pix, fenc(z));
    atomicMax(bev + base + (2u << 20) + pix, fenc(z));
    atomicMax(bev + base + (3u << 20) + pix, fenc(it));
}

__global__ void bev_finalize_kernel(unsigned* __restrict__ bev, int n) {
    int i = blockIdx.x * blockDim.x + threadIdx.x;
    if (i >= n) return;
    int c = (i >> 20) & 3;
    unsigned u = bev[i];
    float f;
    if (c == 0) {
        float cnt = __uint_as_float(u);
        f = ((cnt == 0.0f) ? 1.0f : cnt) * (1.0f / 50.0f);
    } else {
        f = fdec(u);
    }
    ((float*)bev)[i] = f;
}

// ---------------------------------------------------------------------------
// Fused conv3x3 + bias + BN + ReLU (+ optional 2x2 maxpool) as implicit GEMM
// on v_wmma_f32_16x16x32_f16.
//
// Per workgroup (128 threads = 4 waves, 64 output pixels along W):
//   1) zero-fill raw f32 tile (free boundary padding), barrier
//   2) DMA the in-bounds tile global->LDS with GLOBAL_LOAD_ASYNC_TO_LDS_B32
//      (ASYNCcnt path, no VGPR round-trip); weights staged sync meanwhile;
//      s_wait_asynccnt 0, barrier
//   3) build im2col Bt[col][k] (f32->f16) with wave-uniform k decode, barrier
//   4) K-loop (fully unrolled): 6x ds_load_b128 + 2x WMMA per 32-K step
// Contraction index k = (ky*3+kx)*CIN + ci (CIN = power of two).
// With POOL a wave computes conv rows y0 and y0+1 (shared A fragment) and
// pools 2x2 via fmax + __shfl_xor.
// ---------------------------------------------------------------------------
template <int CIN, int COUT, int H, int W, bool POOL>
__global__ __launch_bounds__(128)
void conv_bn_relu(const float* __restrict__ in,   // [4][CIN][H][W]
                  const float* __restrict__ wgt,  // [COUT][CIN][3][3]
                  const float* __restrict__ bp, const float* __restrict__ gp,
                  const float* __restrict__ bep, const float* __restrict__ mp,
                  const float* __restrict__ vp,
                  float* __restrict__ out) {      // [4][COUT][HOUT][WOUT]
    constexpr int KREAL = CIN * 9;
    constexpr int KPAD  = ((KREAL + 31) / 32) * 32;
    constexpr int L2C   = (CIN == 4) ? 2 : (CIN == 8) ? 3 : (CIN == 16) ? 4 : 5;
    constexpr int RREAL = POOL ? 4 : 3;            // raw rows with real data
    constexpr int NCB   = (COUT + 15) / 16;
    constexpr int RSTR  = 68;                      // raw f32 row stride (floats)
    constexpr int KSTR  = KPAD + 8;                // Bt row stride (halves), 16B-mult
    constexpr int HOUT  = POOL ? H / 2 : H;
    constexpr int WOUT  = POOL ? W / 2 : W;

    __shared__ __attribute__((aligned(16))) _Float16 Alds[16 * KPAD];
    __shared__ __attribute__((aligned(16))) float    RawF[CIN * RREAL * RSTR];
    __shared__ __attribute__((aligned(16))) _Float16 Bt0[64 * KSTR];
    __shared__ __attribute__((aligned(16))) _Float16 Bt1[POOL ? 64 * KSTR : 16];

    const int tid  = threadIdx.x;
    const int bz   = blockIdx.z;
    const int b    = bz / NCB;
    const int cb   = bz % NCB;
    const int x0g  = blockIdx.x * 64;
    const int y0   = POOL ? (blockIdx.y * 2) : blockIdx.y;
    const int ytop = y0 - 1;

    // ---- 1) zero-fill raw tile (covers OOB padding) ----------------------
    for (int t = tid; t < CIN * RREAL * RSTR; t += 128)
        RawF[t] = 0.0f;
    __syncthreads();

    // ---- 2) async DMA of in-bounds elements + sync weight staging --------
    {
        const float* inb = in + (size_t)b * CIN * H * W;   // uniform SGPR base
        for (int t = tid; t < CIN * RREAL * 66; t += 128) {
            int ci = t / (RREAL * 66);
            int rr = t % (RREAL * 66);
            int ry = rr / 66, cx = rr % 66;
            int gy = ytop + ry, gx = x0g - 1 + cx;
            if (gy >= 0 && gy < H && gx >= 0 && gx < W) {
                unsigned voff = (unsigned)(((ci * H + gy) * W + gx) * 4);
                unsigned ldst = (unsigned)(uintptr_t)&RawF[(ci * RREAL + ry) * RSTR + cx];
                asm volatile("global_load_async_to_lds_b32 %0, %1, %2"
                             :: "v"(ldst), "v"(voff), "s"(inb) : "memory");
            }
        }
    }
    for (int t = tid; t < 16 * KPAD; t += 128) {
        int m  = t / KPAD, k = t % KPAD;
        int co = cb * 16 + m;
        int kk = k >> L2C;          // (ky*3+kx)
        int ci = k & (CIN - 1);
        float v = 0.0f;
        if (kk < 9 && co < COUT)
            v = wgt[(co * CIN + ci) * 9 + kk];
        Alds[t] = (_Float16)v;
    }
    asm volatile("s_wait_asynccnt 0" ::: "memory");
    __syncthreads();

    // ---- 3) build im2col Bt[col][k]; k decode is wave-uniform ------------
    {
        const int colS = tid & 63;
        for (int kp = (tid >> 6); kp < KPAD / 2; kp += 2) {
            int ke = 2 * kp;
            int ci0 = ke & (CIN - 1),       kk0 = ke >> L2C;
            int ci1 = (ke + 1) & (CIN - 1), kk1 = (ke + 1) >> L2C;
            unsigned lo = 0, hi = 0;
            if (kk0 < 9) {
                int ky = kk0 / 3, kx = kk0 - 3 * ky;
                const float* p = &RawF[(ci0 * RREAL + ky) * RSTR + colS + kx];
                lo |= (unsigned)h16(p[0]);
                if (POOL) hi |= (unsigned)h16(p[RSTR]);
            }
            if (kk1 < 9) {
                int ky = kk1 / 3, kx = kk1 - 3 * ky;
                const float* p = &RawF[(ci1 * RREAL + ky) * RSTR + colS + kx];
                lo |= (unsigned)h16(p[0]) << 16;
                if (POOL) hi |= (unsigned)h16(p[RSTR]) << 16;
            }
            *(unsigned*)&Bt0[colS * KSTR + ke] = lo;
            if (POOL) *(unsigned*)&Bt1[colS * KSTR + ke] = hi;
        }
    }
    __syncthreads();

    const int wid  = tid >> 5;
    const int lane = tid & 31;
    const int half = lane >> 4;
    const int n    = lane & 15;
    const int xw   = x0g + wid * 16;
    const int colb = wid * 16 + n;

    // Per-lane fragment base pointers (all K-loop offsets are constants).
    const _Float16* pA  = &Alds[n * KPAD + half * 16];
    const _Float16* pB0 = &Bt0[colb * KSTR + half * 16];
    const _Float16* pB1 = &Bt1[colb * KSTR + half * 16];

    v8f acc0 = {};
    v8f acc1 = {};

#pragma unroll
    for (int k0 = 0; k0 < KPAD; k0 += 32) {
        union { v4u q[2]; v16h h; } A, B0, B1;
        // A: VGPRs 0-7 = 16 contiguous halves at k0 + 16h
        A.q[0] = *(const v4u*)(pA + k0);
        A.q[1] = *(const v4u*)(pA + k0 + 8);
        // B: VGPRs 0-7 = 16 contiguous halves at k0 + 16h
        B0.q[0] = *(const v4u*)(pB0 + k0);
        B0.q[1] = *(const v4u*)(pB0 + k0 + 8);
        acc0 = __builtin_amdgcn_wmma_f32_16x16x32_f16(
                   false, A.h, false, B0.h, (short)0, acc0, false, false);
        if (POOL) {
            B1.q[0] = *(const v4u*)(pB1 + k0);
            B1.q[1] = *(const v4u*)(pB1 + k0 + 8);
            acc1 = __builtin_amdgcn_wmma_f32_16x16x32_f16(
                       false, A.h, false, B1.h, (short)0, acc1, false, false);
        }
    }

    // ---- epilogue: folded BN + ReLU (+ 2x2 maxpool) ----------------------
#pragma unroll
    for (int r = 0; r < 8; ++r) {
        int co = cb * 16 + r + 8 * half;   // C/D layout: VGPR r -> M = r + 8*half
        float sc = 0.0f, bi = 0.0f;
        if (co < COUT) {
            sc = gp[co] * rsqrtf(vp[co] + 1e-5f);
            bi = (bp[co] - mp[co]) * sc + bep[co];
        }
        float v0 = fmaxf(acc0[r] * sc + bi, 0.0f);
        if (POOL) {
            float v1 = fmaxf(acc1[r] * sc + bi, 0.0f);
            float vm = fmaxf(v0, v1);
            float vo = __shfl_xor(vm, 1, 32);
            float vq = fmaxf(vm, vo);
            if (((n & 1) == 0) && co < COUT)
                out[((b * COUT + co) * HOUT + (y0 >> 1)) * WOUT + ((xw + n) >> 1)] = vq;
        } else {
            if (co < COUT)
                out[((b * COUT + co) * HOUT + y0) * WOUT + (xw + n)] = v0;
        }
    }
}

// ---------------------------------------------------------------------------
// Inputs: points, batch_size, then (w,b,g,be,m,v) x 4 layers.
// Workspace (floats): bev 16,777,216 | act1 8,388,608 | act2 4,194,304 |
// act3 2,097,152  => 120 MB total.
// ---------------------------------------------------------------------------
extern "C" void kernel_launch(void* const* d_in, const int* in_sizes, int n_in,
                              void* d_out, int out_size, void* d_ws, size_t ws_size,
                              hipStream_t stream) {
    const float* pts = (const float*)d_in[0];
    const int npts = in_sizes[0] / 5;

    const float* w1 = (const float*)d_in[2];
    const float* b1 = (const float*)d_in[3];
    const float* g1 = (const float*)d_in[4];
    const float* e1 = (const float*)d_in[5];
    const float* m1 = (const float*)d_in[6];
    const float* v1 = (const float*)d_in[7];
    const float* w2 = (const float*)d_in[8];
    const float* b2 = (const float*)d_in[9];
    const float* g2 = (const float*)d_in[10];
    const float* e2 = (const float*)d_in[11];
    const float* m2 = (const float*)d_in[12];
    const float* v2 = (const float*)d_in[13];
    const float* w3 = (const float*)d_in[14];
    const float* b3 = (const float*)d_in[15];
    const float* g3 = (const float*)d_in[16];
    const float* e3 = (const float*)d_in[17];
    const float* m3 = (const float*)d_in[18];
    const float* v3 = (const float*)d_in[19];
    const float* w4 = (const float*)d_in[20];
    const float* b4 = (const float*)d_in[21];
    const float* g4 = (const float*)d_in[22];
    const float* e4 = (const float*)d_in[23];
    const float* m4 = (const float*)d_in[24];
    const float* v4 = (const float*)d_in[25];

    unsigned* bev_u = (unsigned*)d_ws;
    float*    bev   = (float*)d_ws;
    float*    act1  = bev  + 16777216;   // 4x8x512x512
    float*    act2  = act1 + 8388608;    // 4x16x256x256
    float*    act3  = act2 + 4194304;    // 4x32x128x128
    float*    outp  = (float*)d_out;     // 4x64x128x128

    const int nbev = 4 * 4 * 1024 * 1024;
    bev_init_kernel<<<nbev / 256, 256, 0, stream>>>(bev_u, nbev);
    bev_scatter_kernel<<<(npts + 255) / 256, 256, 0, stream>>>(pts, bev_u, npts);
    bev_finalize_kernel<<<nbev / 256, 256, 0, stream>>>(bev_u, nbev);

    conv_bn_relu<4, 8, 1024, 1024, true>
        <<<dim3(16, 512, 4), 128, 0, stream>>>(bev, w1, b1, g1, e1, m1, v1, act1);
    conv_bn_relu<8, 16, 512, 512, true>
        <<<dim3(8, 256, 4), 128, 0, stream>>>(act1, w2, b2, g2, e2, m2, v2, act2);
    conv_bn_relu<16, 32, 256, 256, true>
        <<<dim3(4, 128, 8), 128, 0, stream>>>(act2, w3, b3, g3, e3, m3, v3, act3);
    conv_bn_relu<32, 64, 128, 128, false>
        <<<dim3(2, 128, 16), 128, 0, stream>>>(act3, w4, b4, g4, e4, m4, v4, outp);
}
